// HSTULayer_53807350284553
// MI455X (gfx1250) — matrix operand order repaired and checked by reading, not compiled, not run on hardware
//
#include <hip/hip_runtime.h>
#include <cstdint>

#define BB   2
#define SS   2048
#define HH   1024
#define NHH  16
#define HD   64
#define O4   4096           // 4*H
#define BS   (BB*SS)        // 4096 rows

typedef __attribute__((ext_vector_type(16))) __bf16 v16bf;
typedef __attribute__((ext_vector_type(8)))  __bf16 v8bf;
typedef __attribute__((ext_vector_type(4)))  __bf16 v4bf;
typedef __attribute__((ext_vector_type(8)))  float  v8f;
typedef __attribute__((ext_vector_type(4)))  uint32_t v4u;
typedef __attribute__((ext_vector_type(4)))  int      v4i;
typedef __attribute__((ext_vector_type(8)))  int      v8i;

// ---- TDM availability probe (device pass only; host pass lacks amdgcn builtins) ----
#if defined(__has_builtin)
#  if __has_builtin(__builtin_amdgcn_tensor_load_to_lds)
#    define USE_TDM 1
#  endif
#endif
#ifndef USE_TDM
#  define USE_TDM 0
#endif
#if defined(__has_include)
#  if __has_include(<hip/amd_detail/amd_gfx1250_TDM.h>)
#    define TDM_ARGS6 1
#  endif
#endif
#ifndef TDM_ARGS6
#  define TDM_ARGS6 0
#endif

// A-fragment (16x32 bf16): lane = m(0..15) + 16*h ; elements 0..15 = K 16h..16h+15
// (contiguous order, matching the B fragments)
__device__ inline v16bf lds_afrag(const __bf16* rowbase, int h) {
  v8bf lo = *(const v8bf*)(rowbase + 16 * h);
  v8bf hi = *(const v8bf*)(rowbase + 16 * h + 8);
  return __builtin_shufflevector(lo, hi, 0,1,2,3,4,5,6,7,8,9,10,11,12,13,14,15);
}

__device__ inline v8f wmma_bf16(v16bf a, v16bf b, v8f c) {
  return __builtin_amdgcn_wmma_f32_16x16x32_bf16(false, a, false, b, (short)0, c, false, false);
}

__device__ inline float silu_f(float x) { return x / (1.0f + __expf(-x)); }

// LDS byte offset of a generic pointer to __shared__ data (ptrtoint of AS(3) ptr)
__device__ inline uint32_t lds_off(const void* p) {
  return (uint32_t)(uintptr_t)(__attribute__((address_space(3))) const void*)p;
}

// 32B global -> LDS async copy (2 x GLOBAL_LOAD_ASYNC_TO_LDS_B128; ASYNCcnt += 2)
__device__ inline void async_cp32(uint32_t lds, uint64_t gaddr) {
  asm volatile("global_load_async_to_lds_b128 %0, %1, off\n\t"
               "global_load_async_to_lds_b128 %0, %1, off offset:16"
               :: "v"(lds), "v"(gaddr) : "memory");
}

// 64B global -> LDS async copy (4 x b128; ASYNCcnt += 4)
__device__ inline void async_cp64(uint32_t lds, uint64_t gaddr) {
  asm volatile("global_load_async_to_lds_b128 %0, %1, off\n\t"
               "global_load_async_to_lds_b128 %0, %1, off offset:16\n\t"
               "global_load_async_to_lds_b128 %0, %1, off offset:32\n\t"
               "global_load_async_to_lds_b128 %0, %1, off offset:48"
               :: "v"(lds), "v"(gaddr) : "memory");
}

#if USE_TDM
// One TDM 2D tile load: 128 rows x 32 bf16, row stride = strideK elements.
// D# group0: count=1 | lds_addr | global_addr(57b) | type=2.
// D# group1: data_size=2B, tensor_dim0=32, tensor_dim1=128, tile 32x128, stride0=strideK.
__device__ inline void tdm_tile(uint32_t lds, uint64_t gaddr, uint32_t strideK) {
  v4u g0;
  g0[0] = 1u;                                            // count=1, user descriptor
  g0[1] = lds;                                           // lds_addr (bytes)
  g0[2] = (uint32_t)gaddr;                               // global_addr[31:0]
  g0[3] = ((uint32_t)(gaddr >> 32) & 0x01FFFFFFu) | (2u << 30);  // addr[56:32] | type=2
  v8i g1;
  g1[0] = 0x00010000;                                    // data_size=1 (2 bytes)
  g1[1] = (int)(32u << 16);                              // tensor_dim0 = 32
  g1[2] = (int)(128u << 16);                             // tensor_dim1 = 128 (lo16 of dim0 hi = 0)
  g1[3] = (int)(32u << 16);                              // tile_dim0 = 32
  g1[4] = 128;                                           // tile_dim1 = 128, tile_dim2 = 0
  g1[5] = (int)strideK;                                  // tensor_dim0_stride
  g1[6] = 0;
  g1[7] = 0;
  v4i z4 = {0, 0, 0, 0};
#if TDM_ARGS6
  v8i z8 = {0, 0, 0, 0, 0, 0, 0, 0};
  __builtin_amdgcn_tensor_load_to_lds(g0, g1, z4, z4, z8, 0);
#else
  __builtin_amdgcn_tensor_load_to_lds(g0, g1, z4, z4, 0);
#endif
}
#endif

// ---------------- fp32 -> bf16 pack ----------------------------------------------------
__global__ void cvt4_f32_bf16(const float4* __restrict__ in, v4bf* __restrict__ out, int n4) {
  int i = blockIdx.x * 256 + threadIdx.x;
  if (i < n4) {
    float4 f = in[i];
    v4bf o;
    o[0] = (__bf16)f.x; o[1] = (__bf16)f.y; o[2] = (__bf16)f.z; o[3] = (__bf16)f.w;
    out[i] = o;
  }
}

// ---------------- bf16 WMMA GEMM: C[M,N] = A[M,K] * B[N,K]^T (+bias, +residual) -------
// Block tile 128x128x32, 256 threads = 8 waves (2x4), each wave 64x32 (4x2 D-tiles).
// Tiles staged by the Tensor Data Mover (wave 0 issues one descriptor per tile,
// pipelined with s_wait_tensorcnt), falling back to per-lane async-to-LDS copies.
__global__ __launch_bounds__(256)
void gemm_bf16_wmma(const __bf16* __restrict__ A, const __bf16* __restrict__ Bw,
                    const float* __restrict__ bias,
                    __bf16* __restrict__ outBf, float* __restrict__ outF,
                    const float* __restrict__ resid,
                    int K, int N, int mode)
{
  __shared__ __bf16 lA[2][128 * 32];
  __shared__ __bf16 lB[2][128 * 32];
  const int t = threadIdx.x;
  const int wave = t >> 5, lane = t & 31;
  const int m = lane & 15, h = lane >> 4;
  const int waveM = wave >> 2, waveN = wave & 3;
  const int m0 = blockIdx.y * 128, n0 = blockIdx.x * 128;

  __builtin_prefetch(&bias[n0 + waveN * 32], 0, 0);   // global_prefetch_b8

  v8f acc[4][2];
  for (int i = 0; i < 4; i++)
    for (int j = 0; j < 2; j++)
      for (int e = 0; e < 8; e++) acc[i][j][e] = 0.0f;

  const int nt = K >> 5;

#if USE_TDM
  const uint64_t gAbase = (uint64_t)(uintptr_t)(A  + (size_t)m0 * K);
  const uint64_t gBbase = (uint64_t)(uintptr_t)(Bw + (size_t)n0 * K);
  const uint32_t ldsA[2] = { lds_off(&lA[0][0]), lds_off(&lA[1][0]) };
  const uint32_t ldsB[2] = { lds_off(&lB[0][0]), lds_off(&lB[1][0]) };
  if (wave == 0) {                       // tile 0 -> buffer 0 (2 TDM descriptors)
    tdm_tile(ldsA[0], gAbase, (uint32_t)K);
    tdm_tile(ldsB[0], gBbase, (uint32_t)K);
  }
#else
  const int ar = t >> 1;          // row 0..127
  const int ac = (t & 1) * 16;    // 16-elem (32B) half-row
  const uint64_t gA = (uint64_t)(uintptr_t)(A  + (size_t)(m0 + ar) * K + ac);
  const uint64_t gB = (uint64_t)(uintptr_t)(Bw + (size_t)(n0 + ar) * K + ac);
  const uint32_t sA[2] = { lds_off(&lA[0][ar * 32 + ac]), lds_off(&lA[1][ar * 32 + ac]) };
  const uint32_t sB[2] = { lds_off(&lB[0][ar * 32 + ac]), lds_off(&lB[1][ar * 32 + ac]) };
  async_cp32(sA[0], gA);
  async_cp32(sB[0], gB);
#endif

  for (int kt = 0; kt < nt; ++kt) {
    const int cur = kt & 1;
    const uint64_t koff = (uint64_t)(kt + 1) * 64;     // 32 bf16 = 64 bytes
#if USE_TDM
    if (wave == 0) {
      if (kt + 1 < nt) {
        tdm_tile(ldsA[cur ^ 1], gAbase + koff, (uint32_t)K);
        tdm_tile(ldsB[cur ^ 1], gBbase + koff, (uint32_t)K);
        __builtin_amdgcn_s_wait_tensorcnt(2);          // tile kt done, kt+1 in flight
      } else {
        __builtin_amdgcn_s_wait_tensorcnt(0);
      }
    }
#else
    if (kt + 1 < nt) {
      async_cp32(sA[cur ^ 1], gA + koff);
      async_cp32(sB[cur ^ 1], gB + koff);
      asm volatile("s_wait_asynccnt 4" ::: "memory");
    } else {
      asm volatile("s_wait_asynccnt 0" ::: "memory");
    }
#endif
    __syncthreads();

    v16bf af[4], bfr[2];
    for (int i = 0; i < 4; i++)
      af[i] = lds_afrag(&lA[cur][(waveM * 64 + 16 * i + m) * 32], h);
    for (int j = 0; j < 2; j++)
      bfr[j] = *(const v16bf*)(&lB[cur][(waveN * 32 + 16 * j + m) * 32 + 16 * h]);

    for (int i = 0; i < 4; i++)
      for (int j = 0; j < 2; j++)
        acc[i][j] = wmma_bf16(af[i], bfr[j], acc[i][j]);
    __syncthreads();   // all waves done with buf[cur] before it is re-filled
  }

  // D layout: VGPR v, lanes 0-15 -> M=v ; lanes 16-31 -> M=8+v ; N = lane%16
  for (int i = 0; i < 4; i++)
    for (int j = 0; j < 2; j++) {
      int gn  = n0 + waveN * 32 + 16 * j + m;
      int gm0 = m0 + waveM * 64 + 16 * i + 8 * h;
      float bcol = bias[gn];
      if (mode == 0) {
        for (int v = 0; v < 8; v++)
          outBf[(size_t)(gm0 + v) * N + gn] = (__bf16)(acc[i][j][v] + bcol);
      } else {
        for (int v = 0; v < 8; v++) {
          size_t off = (size_t)(gm0 + v) * N + gn;
          outF[off] = acc[i][j][v] + bcol + resid[off];
        }
      }
    }
}

// ---------------- GEMM1 epilogue: silu(u), RoPE(q,k), head-major scatter ---------------
__global__ void epilogue1(const __bf16* __restrict__ uvqk,
                          const float* __restrict__ cosb, const float* __restrict__ sinb,
                          __bf16* __restrict__ ub, __bf16* __restrict__ vb,
                          __bf16* __restrict__ qb, __bf16* __restrict__ kb)
{
  int idx = blockIdx.x * 256 + threadIdx.x;      // < BS*O4
  int row = idx >> 12;
  int col = idx & 4095;
  int sect = col >> 10;
  int cc = col & 1023;
  float val = (float)uvqk[idx];
  if (sect == 0) {
    ub[(size_t)row * HH + cc] = (__bf16)silu_f(val);
  } else {
    int head = cc >> 6, d = cc & 63;
    int b = row >> 11, s = row & 2047;
    size_t dst = ((size_t)(b * NHH + head) * SS + s) * HD + d;
    if (sect == 1) {
      vb[dst] = (__bf16)val;
    } else {
      size_t cs = ((size_t)b * SS + s) * HD + d;
      float c = cosb[cs], sn = sinb[cs];
      float partner = (float)uvqk[(size_t)row * O4 + (col + (d < 32 ? 32 : -32))];
      float rot = (d < 32) ? -partner : partner;
      float o = val * c + rot * sn;
      if (sect == 2) qb[dst] = (__bf16)o; else kb[dst] = (__bf16)o;
    }
  }
}

// ---------------- fused silu-attention: scores->silu->mask->PV, flash-style -----------
// Grid (B*NH, S/64); 128 threads = 4 waves; each wave owns 16 query rows.
// K double-buffered via async-to-LDS; next V block's loads issued during compute.
__global__ __launch_bounds__(128)
void attn_silu(const __bf16* __restrict__ qb, const __bf16* __restrict__ kb,
               const __bf16* __restrict__ vb, float* __restrict__ ao)
{
  __shared__ __bf16 Qs[64 * 64];
  __shared__ __bf16 Ks[2][64 * 64];
  __shared__ __bf16 Vt[2][64 * 64];     // transposed: Vt[buf][d][key]
  __shared__ __bf16 Ssc[64 * 64];       // silu(scores), bf16, per-wave 16-row strips
  const int t = threadIdx.x, wave = t >> 5, lane = t & 31;
  const int m = lane & 15, h = lane >> 4;
  const int bh = blockIdx.x;            // b*NH + head
  const int b = bh >> 4, head = bh & 15;
  const int qblk = blockIdx.y, q0 = qblk * 64;

  const int r = t >> 1, seg = (t & 1) * 32;   // 64B strip per thread
  // async-stage Q block (64x64 bf16 = 8KB)
  async_cp64(lds_off(&Qs[r * 64 + seg]),
             (uint64_t)(uintptr_t)(qb + ((size_t)bh * SS + q0) * HD + r * HD + seg));

  v8f oacc[4];
  for (int j = 0; j < 4; j++) for (int e = 0; e < 8; e++) oacc[j][e] = 0.0f;

  const uint32_t sK[2] = { lds_off(&Ks[0][r * 64 + seg]), lds_off(&Ks[1][r * 64 + seg]) };
  const uint64_t kbase = (uint64_t)(uintptr_t)(kb + (size_t)bh * SS * HD + r * HD + seg);
  const uint64_t vbase = (uint64_t)(uintptr_t)(vb + (size_t)bh * SS * HD + r * HD + seg);

  // prologue: K block 0 async; V block 0 into registers
  async_cp64(sK[0], kbase);
  uint4 vr0, vr1, vr2, vr3;
  { const uint4* sv = (const uint4*)vbase; vr0 = sv[0]; vr1 = sv[1]; vr2 = sv[2]; vr3 = sv[3]; }

  for (int kbk = 0; kbk <= qblk; ++kbk) {
    const int cur = kbk & 1;
    { // commit this block's V registers transposed into Vt[cur]
      __bf16 tmp[32];
      *(uint4*)&tmp[0] = vr0; *(uint4*)&tmp[8]  = vr1;
      *(uint4*)&tmp[16] = vr2; *(uint4*)&tmp[24] = vr3;
      for (int d2 = 0; d2 < 32; ++d2) Vt[cur][(seg + d2) * 64 + r] = tmp[d2];
    }
    if (kbk + 1 <= qblk) {
      const uint64_t boff = (uint64_t)(kbk + 1) * (64 * HD * 2);   // 8KB per block
      async_cp64(sK[cur ^ 1], kbase + boff);
      asm volatile("s_wait_asynccnt 4" ::: "memory");  // K[kbk] done, K[kbk+1] in flight
      const uint4* sv = (const uint4*)(vbase + boff);  // next V -> regs during compute
      vr0 = sv[0]; vr1 = sv[1]; vr2 = sv[2]; vr3 = sv[3];
    } else {
      asm volatile("s_wait_asynccnt 0" ::: "memory");
    }
    __syncthreads();

    // scores: 16 q-rows x 64 keys (contract over HD=64 = 2 WMMA k-steps)
    v8f sacc[4];
    for (int j = 0; j < 4; j++) for (int e = 0; e < 8; e++) sacc[j][e] = 0.0f;
    for (int ks = 0; ks < 2; ++ks) {
      v16bf aq = lds_afrag(&Qs[(wave * 16 + m) * 64 + 32 * ks], h);
      for (int j = 0; j < 4; j++) {
        v16bf bk = *(const v16bf*)(&Ks[cur][(16 * j + m) * 64 + 32 * ks + 16 * h]);
        sacc[j] = wmma_bf16(aq, bk, sacc[j]);
      }
    }
    // scale 1/sqrt(64), silu, causal mask; D-layout -> LDS for A-layout re-read
    for (int j = 0; j < 4; j++)
      for (int v = 0; v < 8; v++) {
        int qrow = q0 + wave * 16 + v + 8 * h;
        int kcol = kbk * 64 + 16 * j + m;
        float sv = sacc[j][v] * 0.125f;
        float si = silu_f(sv);
        if (kcol > qrow) si = 0.0f;
        Ssc[(wave * 16 + v + 8 * h) * 64 + 16 * j + m] = (__bf16)si;
      }
    // PV: out(16x64) += attn(16x64) * V(64x64); wave reads only its own Ssc strip
    for (int ks = 0; ks < 2; ++ks) {
      v16bf aa = lds_afrag(&Ssc[(wave * 16 + m) * 64 + 32 * ks], h);
      for (int j2 = 0; j2 < 4; j2++) {
        v16bf bv = *(const v16bf*)(&Vt[cur][(16 * j2 + m) * 64 + 32 * ks + 16 * h]);
        oacc[j2] = wmma_bf16(aa, bv, oacc[j2]);
      }
    }
    __syncthreads();
  }

  for (int j2 = 0; j2 < 4; j2++)
    for (int v = 0; v < 8; v++) {
      int srow = q0 + wave * 16 + v + 8 * h;
      int d = 16 * j2 + m;
      ao[((size_t)b * SS + srow) * HH + head * HD + d] = oacc[j2][v];
    }
}

// ---------------- RMS-norm * gate_w * silu(u) -> bf16 ---------------------------------
__global__ __launch_bounds__(256)
void rmsnorm_gate(const float* __restrict__ ao, const float* __restrict__ gw,
                  const __bf16* __restrict__ ub, __bf16* __restrict__ gated)
{
  __shared__ float red[256];
  int row = blockIdx.x, t = threadIdx.x;
  const float* r = ao + (size_t)row * HH;
  float ss = 0.0f;
  for (int c = t; c < HH; c += 256) { float x = r[c]; ss += x * x; }
  red[t] = ss; __syncthreads();
  for (int s = 128; s > 0; s >>= 1) {
    if (t < s) red[t] += red[t + s];
    __syncthreads();
  }
  float rs = rsqrtf(red[0] / (float)HH + 1e-6f);
  for (int c = t; c < HH; c += 256) {
    float g = gw[c] * r[c] * rs * (float)ub[(size_t)row * HH + c];
    gated[(size_t)row * HH + c] = (__bf16)g;
  }
}

// ---------------- launch ----------------------------------------------------------------
extern "C" void kernel_launch(void* const* d_in, const int* in_sizes, int n_in,
                              void* d_out, int out_size, void* d_ws, size_t ws_size,
                              hipStream_t stream)
{
  (void)in_sizes; (void)n_in; (void)out_size; (void)ws_size;
  const float* x      = (const float*)d_in[0];
  const float* cosb   = (const float*)d_in[1];
  const float* sinb   = (const float*)d_in[2];
  /* d_in[3] attn_mask: causal tril of ones -> handled implicitly */
  const float* W_uvqk = (const float*)d_in[4];
  const float* b_uvqk = (const float*)d_in[5];
  const float* gate_w = (const float*)d_in[6];
  const float* W_out  = (const float*)d_in[7];
  const float* b_out  = (const float*)d_in[8];
  float* out = (float*)d_out;

  char* ws = (char*)d_ws;
  __bf16* xb    = (__bf16*)(ws);                          //  8 MiB  bf16 x       [BS][H]
  __bf16* wub   = (__bf16*)(ws + (8ull  << 20));          //  8 MiB  bf16 W_uvqk  [4H][H]
  __bf16* wob   = (__bf16*)(ws + (16ull << 20));          //  2 MiB  bf16 W_out   [H][H]
  __bf16* uvqk  = (__bf16*)(ws + (18ull << 20));          // 32 MiB  bf16 uvqk    [BS][4H]
  __bf16* ub    = (__bf16*)(ws + (50ull << 20));          //  8 MiB  silu(u)      [BS][H]
  __bf16* vb    = (__bf16*)(ws + (58ull << 20));          //  8 MiB  v  [B][NH][S][HD]
  __bf16* qb    = (__bf16*)(ws + (66ull << 20));          //  8 MiB  q(rope) head-major
  __bf16* kbuf  = (__bf16*)(ws + (74ull << 20));          //  8 MiB  k(rope) head-major
  float*  ao    = (float*)(ws + (82ull << 20));           // 16 MiB  attn_out f32 [BS][H]
  __bf16* gated = (__bf16*)(ws + (98ull << 20));          //  8 MiB  gated bf16   [BS][H]

  cvt4_f32_bf16<<<dim3(BS * HH / 4 / 256), 256, 0, stream>>>((const float4*)x, (v4bf*)xb, BS * HH / 4);
  cvt4_f32_bf16<<<dim3(O4 * HH / 4 / 256), 256, 0, stream>>>((const float4*)W_uvqk, (v4bf*)wub, O4 * HH / 4);
  cvt4_f32_bf16<<<dim3(HH * HH / 4 / 256), 256, 0, stream>>>((const float4*)W_out, (v4bf*)wob, HH * HH / 4);

  gemm_bf16_wmma<<<dim3(O4 / 128, BS / 128), 256, 0, stream>>>(
      xb, wub, b_uvqk, uvqk, nullptr, nullptr, HH, O4, 0);

  epilogue1<<<dim3(BS * O4 / 256), 256, 0, stream>>>(uvqk, cosb, sinb, ub, vb, qb, kbuf);

  attn_silu<<<dim3(BB * NHH, SS / 64), 128, 0, stream>>>(qb, kbuf, vb, ao);

  rmsnorm_gate<<<dim3(BS), 256, 0, stream>>>(ao, gate_w, ub, gated);

  gemm_bf16_wmma<<<dim3(HH / 128, BS / 128), 256, 0, stream>>>(
      gated, wob, b_out, nullptr, out, x, HH, HH, 1);
}
